// STGCN_Export_Model_v13_33792802684996
// MI455X (gfx1250) — matrix operand order repaired and checked by reading, not compiled, not run on hardware
//
#include <hip/hip_runtime.h>
#include <hip/hip_bf16.h>

typedef __attribute__((ext_vector_type(16))) _Float16 v16h;
typedef __attribute__((ext_vector_type(8)))  float    v8f;

static constexpr int KS_[5]      = {3, 3, 5, 9, 9};
static constexpr int DILS_[5]    = {1, 2, 4, 8, 12};
static constexpr int CB_[5]      = {12, 12, 12, 12, 16};
static constexpr int CBSTART_[5] = {0, 12, 24, 36, 48};
static constexpr int BUFLEN_[5]  = {2, 4, 16, 64, 96};

struct MstcnParams {
  const float* kern[5];
  const float* bias[5];
  const float* wa1; const float* ba1;
  const float* wa2; const float* ba2;
  const float* wo;  const float* bo;
  const float* buf[5];
};

struct Params {
  const float* input_frame;
  const float* A;
  const float* W_gcn;
  MstcnParams m1, m2;
  const float* w_d; const float* b_d;
  float* out;
};

// Fragment-swizzled LDS: the 16 halves lane L needs for fragment `idx` sit
// contiguously at (idx*32 + L)*16 -> one 32B v16h load (2x ds_load_b128).
__device__ inline v16h load_frag_sw(const _Float16* src, int idx) {
  const int lane = threadIdx.x & 31;
  return *(const v16h*)(src + (idx * 32 + lane) * 16);
}

// ---- One MS-TCN block ----
__device__ inline void mstcn_layer(const MstcnParams& p, const float* xin, float* yout,
                                   _Float16* mrowsw, _Float16* wohsw,
                                   float* merged, float* a1v) {
  const int tid = threadIdx.x, lane = tid & 31, wave = tid >> 5;

  // zero A-side fragments for the gated row (2 k-frags x 32 slots)
  if (tid < 64) *(v16h*)(mrowsw + tid * 16) = v16h{};
  // wo -> f16 LDS, packed directly in B-swizzled fragment order (8 frags x 32 slots)
  {
    const int frag = tid >> 5, kf = frag >> 2, ct = frag & 3;
    const int g = lane >> 4, col = lane & 15;
    const int n = ct * 16 + col;
    v16h b;
#pragma unroll
    for (int j = 0; j < 16; ++j) {
      const int k = kf * 32 + g * 16 + j;
      b[j] = (_Float16)p.wo[k * 64 + n];
    }
    *(v16h*)(wohsw + tid * 16) = b;
  }

  // Dilated conv branches: thread t computes merged[t]
  if (tid < 64) {
    const int b = (tid < 48) ? (tid / 12) : 4;
    const int o = tid - CBSTART_[b];
    const int K = KS_[b], d = DILS_[b], cb = CB_[b];
    const float* kern = p.kern[b];
    const float* buf  = p.buf[b];
    float acc = p.bias[b][o];
    for (int k = 0; k < K; ++k) {
      const float* src = (k == K - 1) ? xin : (buf + (k * d) * 64);
      const float* kr  = kern + (k * 64) * cb + o;
      for (int c = 0; c < 64; ++c)
        acc = fmaf(src[c], kr[c * cb], acc);
    }
    merged[tid] = fmaxf(acc, 0.f);
  }
  __syncthreads();

  if (tid < 16) {
    float acc = p.ba1[tid];
    for (int c = 0; c < 64; ++c) acc = fmaf(merged[c], p.wa1[c * 16 + tid], acc);
    a1v[tid] = fmaxf(acc, 0.f);
  }
  __syncthreads();

  if (tid < 64) {
    float z = p.ba2[tid];
    for (int j = 0; j < 16; ++j) z = fmaf(a1v[j], p.wa2[j * 64 + tid], z);
    const float att = 1.f / (1.f + __expf(-z));
    // gated value -> A-swizzled row 0, k = tid
    const int kf = tid >> 5, kk = tid & 31;
    const int g = kk >> 4, j = kk & 15;
    mrowsw[((kf * 32) + g * 16 + 0) * 16 + j] = (_Float16)(merged[tid] * att);
  }
  __syncthreads();

  // y = (merged*att) @ wo + bo : 4 col-tiles x 2 K-steps
  if (wave < 4) {
    v8f acc = {};
#pragma unroll
    for (int kf = 0; kf < 2; ++kf) {
      v16h af = load_frag_sw(mrowsw, kf);
      v16h bf = load_frag_sw(wohsw, kf * 4 + wave);
      acc = __builtin_amdgcn_wmma_f32_16x16x32_f16(false, af, false, bf, (short)0, acc,
                                                   false, false);
    }
    if (lane < 16) yout[wave * 16 + lane] = acc[0] + p.bo[wave * 16 + lane];
  }
  __syncthreads();
}

__global__ void __launch_bounds__(256)
stgcn_kernel(Params P) {
  __shared__ __align__(32) _Float16 Xsw[2 * 32 * 16];   // X, A-swizzled (2 row-tiles)
  __shared__ __align__(32) _Float16 Asw[2 * 32 * 16];   // adjacency, A-swizzled
  __shared__ __align__(32) _Float16 Wsw[4 * 32 * 16];   // W_gcn, B-swizzled (4 col-frags)
  __shared__ __align__(32) _Float16 XWsw[4 * 32 * 16];  // X@W, B-swizzled
  __shared__ __align__(32) _Float16 mrowsw[2 * 32 * 16];
  __shared__ __align__(32) _Float16 wohsw[8 * 32 * 16];
  __shared__ float Hs[32 * 64];
  __shared__ float pooled[64];
  __shared__ float merged[64];
  __shared__ float a1v[16];
  __shared__ float y1[64];
  __shared__ float y2[64];

  const int tid = threadIdx.x, lane = tid & 31, wave = tid >> 5;

  // warm L2 for the weights while we stage LDS (global_prefetch_b8)
  if (tid == 0) {
    __builtin_prefetch(P.m1.wo, 0, 1);
    __builtin_prefetch(P.m2.wo, 0, 1);
    __builtin_prefetch(P.m1.wa1, 0, 1);
    __builtin_prefetch(P.m2.wa1, 0, 1);
    __builtin_prefetch(P.m1.kern[4], 0, 1);
    __builtin_prefetch(P.m2.kern[4], 0, 1);
    __builtin_prefetch(P.w_d, 0, 1);
  }

  // X (21x18 padded to 32x32) in A-swizzle: pair p = tile*32 + slot
  if (tid < 64) {
    const int tile = tid >> 5, slot = tid & 31;
    const int row = tile * 16 + (slot & 15), g = slot >> 4;
    v16h a;
#pragma unroll
    for (int j = 0; j < 16; ++j) {
      const int k = g * 16 + j;
      float v = 0.f;
      if (row == 0)       { if (k < 18) v = P.input_frame[k]; }
      else if (row <= 20) { if (k < 10) v = P.input_frame[18 + (row - 1) * 10 + k]; }
      a[j] = (_Float16)v;
    }
    *(v16h*)(Xsw + tid * 16) = a;
  }
  // adjacency (21x21 padded to 32x32) in A-swizzle
  else if (tid < 128) {
    const int p = tid - 64;
    const int tile = p >> 5, slot = p & 31;
    const int row = tile * 16 + (slot & 15), g = slot >> 4;
    v16h a;
#pragma unroll
    for (int j = 0; j < 16; ++j) {
      const int k = g * 16 + j;
      a[j] = (_Float16)((row < 21 && k < 21) ? P.A[row * 21 + k] : 0.f);
    }
    *(v16h*)(Asw + p * 16) = a;
  }
  // W_gcn (18x64 padded to 32x64) in B-swizzle: 4 frags x 32 slots
  else {
    const int p = tid - 128;
    const int frag = p >> 5, slot = p & 31;
    const int g = slot >> 4, col = slot & 15;
    const int n = frag * 16 + col;
    v16h b;
#pragma unroll
    for (int j = 0; j < 16; ++j) {
      const int k = g * 16 + j;
      b[j] = (_Float16)((k < 18) ? P.W_gcn[k * 64 + n] : 0.f);
    }
    *(v16h*)(Wsw + p * 16) = b;
  }
  __syncthreads();

  // XW = X @ W : 8 waves, one 16x16 (K=32) tile each; write D in B-swizzle for next matmul
  {
    const int rt = wave >> 2, ct = wave & 3;
    v16h af = load_frag_sw(Xsw, rt);
    v16h bf = load_frag_sw(Wsw, ct);
    v8f acc = {};
    acc = __builtin_amdgcn_wmma_f32_16x16x32_f16(false, af, false, bf, (short)0, acc,
                                                 false, false);
    const int col = lane & 15, g = lane >> 4;
#pragma unroll
    for (int v = 0; v < 8; ++v) {
      const int M = rt * 16 + v + g * 8;          // K index of next matmul
      const int gb = (M >> 4) & 1, jb = M & 15;
      XWsw[((ct * 32) + gb * 16 + col) * 16 + jb] = (_Float16)acc[v];
    }
  }
  __syncthreads();

  // H = relu(A @ XW) : 8 waves, one tile each
  {
    const int rt = wave >> 2, ct = wave & 3;
    v16h af = load_frag_sw(Asw, rt);
    v16h bf = load_frag_sw(XWsw, ct);
    v8f acc = {};
    acc = __builtin_amdgcn_wmma_f32_16x16x32_f16(false, af, false, bf, (short)0, acc,
                                                 false, false);
    const int col = lane & 15, g = lane >> 4;
#pragma unroll
    for (int v = 0; v < 8; ++v)
      Hs[(rt * 16 + v + g * 8) * 64 + ct * 16 + col] = fmaxf(acc[v], 0.f);
  }
  __syncthreads();

  // mean over the 21 real nodes
  if (tid < 64) {
    float s = 0.f;
    for (int n = 0; n < 21; ++n) s += Hs[n * 64 + tid];
    pooled[tid] = s * (1.f / 21.f);
  }
  __syncthreads();

  mstcn_layer(P.m1, pooled, y1, mrowsw, wohsw, merged, a1v);
  mstcn_layer(P.m2, y1,     y2, mrowsw, wohsw, merged, a1v);

  // dense head: tanh(y2 @ w_d + b_d)
  if (tid < 32) {
    float acc = P.b_d[tid];
    for (int c = 0; c < 64; ++c) acc = fmaf(y2[c], P.w_d[c * 32 + tid], acc);
    P.out[tid] = tanhf(acc);
  }

  // shifted state buffers: new_buf = [buf rows 1..L-1, layer input]
  {
    int off = 32;
    for (int li = 0; li < 2; ++li) {
      const MstcnParams& m = li ? P.m2 : P.m1;
      for (int b = 0; b < 5; ++b) {
        const int L = BUFLEN_[b];
        for (int i = tid; i < L * 64; i += 256) {
          const int row = i >> 6, c = i & 63;
          float v;
          if (row < L - 1) v = m.buf[b][(row + 1) * 64 + c];
          else             v = li ? y1[c] : pooled[c];
          P.out[off + i] = v;
        }
        off += L * 64;
      }
    }
  }
}

extern "C" void kernel_launch(void* const* d_in, const int* in_sizes, int n_in,
                              void* d_out, int out_size, void* d_ws, size_t ws_size,
                              hipStream_t stream) {
  (void)in_sizes; (void)out_size; (void)d_ws; (void)ws_size;
  Params P{};
  auto F = [&](int i) { return (const float*)d_in[i]; };

  P.input_frame = F(0);
  for (int b = 0; b < 5; ++b) P.m1.buf[b] = F(1 + b);
  for (int b = 0; b < 5; ++b) P.m2.buf[b] = F(6 + b);
  P.A = F(11); P.W_gcn = F(12);

  // t*_kernels / t*_biases are python lists: either each element is its own input
  // (n_in == 47) or they arrive concatenated flat (n_in == 31).
  const bool split = (n_in >= 47);
  static const int KOFF[5] = {0, 2304, 4608, 8448, 15360}; // cumulative k*64*cb
  int idx = 13;
  for (int li = 0; li < 2; ++li) {
    MstcnParams& m = li ? P.m2 : P.m1;
    if (split) {
      for (int b = 0; b < 5; ++b) m.kern[b] = F(idx++);
      for (int b = 0; b < 5; ++b) m.bias[b] = F(idx++);
    } else {
      const float* kflat = F(idx++);
      const float* bflat = F(idx++);
      for (int b = 0; b < 5; ++b) {
        m.kern[b] = kflat + KOFF[b];
        m.bias[b] = bflat + CBSTART_[b];
      }
    }
    m.wa1 = F(idx++); m.ba1 = F(idx++);
    m.wa2 = F(idx++); m.ba2 = F(idx++);
    m.wo  = F(idx++); m.bo  = F(idx++);
  }
  P.w_d = F(idx++); P.b_d = F(idx++);
  P.out = (float*)d_out;

  stgcn_kernel<<<dim3(1), dim3(256), 0, stream>>>(P);
}
